// BottomUpGNN_81853486727258
// MI455X (gfx1250) — matrix operand order repaired and checked by reading, not compiled, not run on hardware
//
#include <hip/hip_runtime.h>
#include <hip/hip_bf16.h>
#include <math.h>
#include <initializer_list>

// ---------------------------------------------------------------------------
// BottomUpGNN for MI455X (gfx1250, wave32, WMMA).
//
// Heavy compute = dense MLPs (~130 GFLOP) -> v_wmma_f32_16x16x32_bf16,
// 8 WMMAs per K-step per wave (128x128 block tile, 32x64 per wave).
// Heavy memory  = edge gather/scatter (~1GB/pass) -> global_atomic_add_f32.
// All row counts are multiples of 128, K multiples of 32, N in {128,256},
// so the WMMA GEMM runs guard-free (EXEC all ones, as the ISA requires).
// B tile is stored in LDS in fragment-ready order so every fragment load is
// ds_load_b128 (no VALU shuffling to assemble v16bf operands).
// ---------------------------------------------------------------------------

typedef __attribute__((ext_vector_type(16))) __bf16 v16bf;
typedef __attribute__((ext_vector_type(8)))  float  v8f;

#define BLK 256      // 8 waves of 32
#define BM  128
#define BN  128
#define BK  32
#define LDA 20       // dword stride of an A-tile row (16 data + 4 pad, keeps 16B align)
#define LBSZ 2176    // B tile: 128 cols x 2 groups x 8 dwords + skew

__device__ __forceinline__ unsigned f2bf_pair(float lo, float hi) {
  // two f32 -> packed bf16 pair (round-to-nearest-even), low half = even K
  union { float f; unsigned u; } a, b;
  a.f = lo; b.f = hi;
  unsigned ua = a.u + 0x7FFFu + ((a.u >> 16) & 1u);
  unsigned ub = b.u + 0x7FFFu + ((b.u >> 16) & 1u);
  return (ua >> 16) | (ub & 0xFFFF0000u);
}

__device__ __forceinline__ float gelu_f(float x) {
  return 0.5f * x * (1.0f + erff(x * 0.70710678118654752f));  // exact GELU
}

// B chunk base for (column c, lane-group hig): 8 contiguous dwords, 16B aligned,
// skewed every 4 columns to spread LDS banks.
__device__ __forceinline__ int b_chunk(int c, int hig) {
  return (c * 2 + hig) * 8 + (c >> 2) * 4;
}

union FragU { unsigned u[8]; v16bf v; };

// C[M,N] = act( concat_K(A1[M,K1], A2[M,K2]) @ W[K,N] + bias ), optionally
// gated per-row by (gate[row] > 0) -> skip store (implements jnp.where select
// when C aliases the old xs buffer).
__global__ __launch_bounds__(BLK)
void wmma_gemm_kernel(const float* __restrict__ A1, int K1,
                      const float* __restrict__ A2, int K2,
                      const float* __restrict__ Wt, const float* __restrict__ bias,
                      float* __restrict__ C, const float* __restrict__ gate,
                      int M, int N, int act)
{
  __shared__ unsigned smem[BM * LDA + LBSZ];
  unsigned* lA = smem;
  unsigned* lB = smem + BM * LDA;

  const int K    = K1 + K2;
  const int tid  = threadIdx.x;
  const int lane = tid & 31;
  const int wave = tid >> 5;
  const int wm   = wave & 3;          // 4 waves along M (32 rows each)
  const int wn   = wave >> 2;         // 2 waves along N (64 cols each)
  const int lrow = lane & 15;
  const int hig  = lane >> 4;         // lane group 0/1
  const int rowBase = blockIdx.y * BM;
  const int colBase = blockIdx.x * BN;
  (void)M;

  v8f acc[2][4];
#pragma unroll
  for (int mi = 0; mi < 2; ++mi)
#pragma unroll
    for (int ni = 0; ni < 4; ++ni)
      acc[mi][ni] = (v8f){0.f, 0.f, 0.f, 0.f, 0.f, 0.f, 0.f, 0.f};

  for (int k0 = 0; k0 < K; k0 += BK) {
    // ---- cooperative load: A tile BMxBK f32 -> bf16 pairs in LDS ----
#pragma unroll
    for (int i = 0; i < 8; ++i) {
      int d  = tid + i * BLK;         // 2048 dwords
      int r  = d >> 4;
      int kp = d & 15;
      int k  = k0 + kp * 2;
      int gr = rowBase + r;
      float lo, hi2;
      if (k < K1) { const float* p = A1 + (size_t)gr * K1 + k;        lo = p[0]; hi2 = p[1]; }
      else        { const float* p = A2 + (size_t)gr * K2 + (k - K1); lo = p[0]; hi2 = p[1]; }
      lA[r * LDA + kp] = f2bf_pair(lo, hi2);
    }
    // ---- cooperative load: W tile BKxBN -> fragment-ready bf16 pairs ----
#pragma unroll
    for (int i = 0; i < 8; ++i) {
      int d  = tid + i * BLK;         // 2048 dwords
      int kp = d >> 7;                // K-pair row 0..15
      int n  = d & 127;
      int gk = k0 + kp * 2;
      size_t base = (size_t)gk * N + colBase + n;
      lB[b_chunk(n, kp >> 3) + (kp & 7)] = f2bf_pair(Wt[base], Wt[base + N]);
    }
    if (k0 + BK < K)  // speculative prefetch of next weight tile -> global_prefetch
      __builtin_prefetch(Wt + (size_t)(k0 + BK) * N + colBase + (tid & 127), 0, 1);
    __syncthreads();

    // ---- fragments in contiguous K order (k = 16*hig + s), same for A and B ----
    v16bf afrag[2];
#pragma unroll
    for (int mi = 0; mi < 2; ++mi) {
      int r = wm * 32 + mi * 16 + lrow;
      const unsigned* p = &lA[r * LDA + hig * 8];  // 16B aligned (LDA%4==0)
      FragU ua;
      *(uint4*)&ua.u[0] = *(const uint4*)&p[0];    // V0..V3: K {0..7 | 16..23}
      *(uint4*)&ua.u[4] = *(const uint4*)&p[4];    // V4..V7: K {8..15 | 24..31}
      afrag[mi] = ua.v;
    }
#pragma unroll
    for (int ni = 0; ni < 4; ++ni) {
      int c = wn * 64 + ni * 16 + lrow;
      const unsigned* q = &lB[b_chunk(c, hig)];    // lanes 0-15: K0..15, 16-31: K16..31
      FragU ub;
      *(uint4*)&ub.u[0] = *(const uint4*)&q[0];
      *(uint4*)&ub.u[4] = *(const uint4*)&q[4];
#pragma unroll
      for (int mi = 0; mi < 2; ++mi)
        acc[mi][ni] = __builtin_amdgcn_wmma_f32_16x16x32_bf16(
            false, afrag[mi], false, ub.v, (short)0, acc[mi][ni], false, false);
    }
    __syncthreads();
  }

  // ---- epilogue: bias (+GELU) (+indeg gating), C/D layout per ISA ----
#pragma unroll
  for (int mi = 0; mi < 2; ++mi) {
    int rbase = rowBase + wm * 32 + mi * 16 + hig * 8;
#pragma unroll
    for (int v = 0; v < 8; ++v) {
      int row = rbase + v;
      if (gate && gate[row] <= 0.0f) continue;     // keep old xs value
#pragma unroll
      for (int ni = 0; ni < 4; ++ni) {
        int col = colBase + wn * 64 + ni * 16 + lrow;
        float x = acc[mi][ni][v] + (bias ? bias[col] : 0.0f);
        if (act) x = gelu_f(x);
        C[(size_t)row * N + col] = x;
      }
    }
  }
}

// agg[dst,:] += xs[src,:] for edges whose dst depth == d_t; indeg[dst] += 1
__global__ __launch_bounds__(256)
void scatter_add_kernel(const int* __restrict__ edges, const float* __restrict__ xsrc,
                        float* __restrict__ agg, float* __restrict__ indeg,
                        const int* __restrict__ depth_dst, int d_t, int E)
{
  int e = blockIdx.x * 2 + (threadIdx.x >> 7);
  int f = threadIdx.x & 127;
  if (e >= E) return;
  int dst = edges[E + e];
  if (depth_dst[dst] != d_t) return;
  int src = edges[e];
  atomicAdd(&agg[(size_t)dst * 128 + f], xsrc[(size_t)src * 128 + f]);
  if (f == 0) atomicAdd(&indeg[dst], 1.0f);
}

__global__ __launch_bounds__(256)
void zero_kernel(float* __restrict__ p, long n)
{
  long i = (long)blockIdx.x * 256 + threadIdx.x;
  long stride = (long)gridDim.x * 256;
  for (; i < n; i += stride) p[i] = 0.0f;
}

// pooled[b,d] = mean_r( xs_rel[b*32+r, d] / depth_rel[b*32+r] )
__global__ __launch_bounds__(128)
void pool_rel_kernel(const float* __restrict__ xs_rel, const int* __restrict__ depth_rel,
                     float* __restrict__ pooled)
{
  int b = blockIdx.x, d = threadIdx.x;
  float s = 0.0f;
#pragma unroll 4
  for (int r = 0; r < 32; ++r) {
    int n = b * 32 + r;
    s += xs_rel[(size_t)n * 128 + d] / (float)depth_rel[n];
  }
  pooled[(size_t)b * 128 + d] = s * (1.0f / 32.0f);
}

// out[row] = h[row,:] . W[:,1] + b  (final classifier layer, N=1)
__global__ __launch_bounds__(128)
void final_fc_kernel(const float* __restrict__ h, const float* __restrict__ W,
                     const float* __restrict__ b, float* __restrict__ out)
{
  __shared__ float red[128];
  int row = blockIdx.x, t = threadIdx.x;
  red[t] = h[(size_t)row * 128 + t] * W[t];
  __syncthreads();
  for (int s = 64; s > 0; s >>= 1) {
    if (t < s) red[t] += red[t + s];
    __syncthreads();
  }
  if (t == 0) out[row] = red[0] + b[0];
}

// ---------------------------------------------------------------------------
extern "C" void kernel_launch(void* const* d_in, const int* in_sizes, int n_in,
                              void* d_out, int out_size, void* d_ws, size_t ws_size,
                              hipStream_t stream)
{
  (void)out_size; (void)ws_size;
  // types: 0=Table 1=Field 2=OP 3=Rel 4=Literal
  const float* X[5]     = { (const float*)d_in[0], (const float*)d_in[2], (const float*)d_in[4],
                            (const float*)d_in[6], (const float*)d_in[8] };
  const int*   DEPTH[5] = { (const int*)d_in[1], (const int*)d_in[3], (const int*)d_in[5],
                            (const int*)d_in[7], (const int*)d_in[9] };
  static const int NC[5]  = { 8192, 32768, 65536, 32768, 32768 };
  static const int IND[5] = { 64, 96, 128, 160, 32 };
  const int E = 131072;

  // --- locate param leaves (two plausible pytree orders; detect by size) ---
  int hs[5], os[5], cs;
  bool sorted_layout = (n_in > 19) && (in_sizes[19] == 128);  // classifier b0 first
  if (sorted_layout) {  // jax tree_flatten: dict keys sorted
    cs = 18;                                           // Wc0,bc0,Wc1,bc1
    const int order[5] = {1, 4, 2, 3, 0};              // Field,Literal,OP,Rel,Table
    for (int i = 0; i < 5; ++i) { hs[order[i]] = 22 + 4 * i; os[order[i]] = 42 + 4 * i; }
  } else {              // dict insertion order: hidden, out, classifier
    for (int i = 0; i < 5; ++i) { hs[i] = 18 + 4 * i; os[i] = 38 + 4 * i; }
    cs = 58;
  }
  auto Pf = [&](int i) { return (const float*)d_in[i]; };

  // --- workspace carve (floats) ---
  float* w = (float*)d_ws;
  size_t off = 0;
  auto alloc = [&](size_t n) { float* p = w + off; off += n; return p; };
  float* xs[5];
  for (int t = 0; t < 5; ++t) xs[t] = alloc((size_t)NC[t] * 128);
  float* agg[5] = {}; float* indeg[5] = {};
  agg[1] = alloc((size_t)32768 * 128);   // Field / OP / Rel contiguous -> one zero pass
  agg[2] = alloc((size_t)65536 * 128);
  agg[3] = alloc((size_t)32768 * 128);
  indeg[1] = alloc(32768); indeg[2] = alloc(65536); indeg[3] = alloc(32768);
  float* tmp    = alloc((size_t)65536 * 256);  // MLP hidden activations (largest type)
  float* pooled = alloc((size_t)1024 * 128);
  float* clsh   = alloc((size_t)1024 * 128);
  const long ZN = (long)131072 * 128 + 131072; // agg + indeg region

  auto gemm = [&](const float* A1, int K1, const float* A2, int K2,
                  const float* W, const float* b, float* C, const float* gate,
                  int M, int N, int act) {
    dim3 g((unsigned)(N / BN), (unsigned)(M / BM));
    wmma_gemm_kernel<<<g, BLK, 0, stream>>>(A1, K1, A2, K2, W, b, C, gate, M, N, act);
  };

  // 1) per-type hidden MLPs: x -> GELU(x@W0+b0)@W1+b1 -> xs[t]  [Nt,128]
  for (int t = 0; t < 5; ++t) {
    gemm(X[t], IND[t], X[t], 0, Pf(hs[t] + 0), Pf(hs[t] + 1), tmp, nullptr, NC[t], 256, 1);
    gemm(tmp, 256, tmp, 0,     Pf(hs[t] + 2), Pf(hs[t] + 3), xs[t], nullptr, NC[t], 128, 0);
  }

  // 2) depth-ordered bottom-up passes (dst depth 4,3,2,1)
  struct EdgeSpec { int idx, st, dt; };
  static const EdgeSpec EG[8] = { {10,3,3},{11,3,2},{12,2,3},{13,2,2},
                                  {14,1,2},{15,1,3},{16,0,1},{17,4,2} };
  for (int s = 0; s < 4; ++s) {
    int dts = 4 - s;
    zero_kernel<<<2048, 256, 0, stream>>>(agg[1], ZN);
    for (int i = 0; i < 8; ++i)
      scatter_add_kernel<<<E / 2, 256, 0, stream>>>(
          (const int*)d_in[EG[i].idx], xs[EG[i].st], agg[EG[i].dt], indeg[EG[i].dt],
          DEPTH[EG[i].dt], dts, E);
    for (int t : {1, 2, 3}) {  // Field, OP, Rel
      // upd = MLP(concat([agg, xs])) ; xs = indeg>0 ? upd : xs  (gated store)
      gemm(agg[t], 128, xs[t], 128, Pf(os[t] + 0), Pf(os[t] + 1), tmp, nullptr, NC[t], 256, 1);
      gemm(tmp, 256, tmp, 0,        Pf(os[t] + 2), Pf(os[t] + 3), xs[t], indeg[t], NC[t], 128, 0);
    }
  }

  // 3) depth-weighted mean-pool of Rel per graph, then classifier
  pool_rel_kernel<<<1024, 128, 0, stream>>>(xs[3], DEPTH[3], pooled);
  gemm(pooled, 128, pooled, 0, Pf(cs + 0), Pf(cs + 1), clsh, nullptr, 1024, 128, 1);
  final_fc_kernel<<<1024, 128, 0, stream>>>(clsh, Pf(cs + 2), Pf(cs + 3), (float*)d_out);
}
